// TopologyAwareAttention_34316788695082
// MI455X (gfx1250) — matrix operand order repaired and checked by reading, not compiled, not run on hardware
//
#include <hip/hip_runtime.h>
#include <hip/hip_bf16.h>

typedef __attribute__((ext_vector_type(16))) _Float16 v16h;
typedef __attribute__((ext_vector_type(8)))  float    v8f;

#define B   2
#define NQ  900
#define HW  256
#define E   256
#define NH  8

// workspace layout (bytes)
#define OFF_IDX   0                        // int[B*NQ]
#define OFF_CNT   (OFF_IDX  + B*NQ*4)      // int[B*256]
#define OFF_MEAN  (OFF_CNT  + B*256*4)     // float[B]
#define OFF_VBAR  (OFF_MEAN + B*4)         // float[256]
#define OFF_CM    (OFF_VBAR + 256*4)       // float[1]
#define OFF_LUT   (OFF_CM   + 4)           // float[B*256]

// ---------------------------------------------------------------------------
// K1: per-(b,q) argmax over 256 cells (first-max tie-break, like jnp.argmax)
//     + histogram of winning cells per batch. One wave32 per (b,q).
//     Each lane scans a contiguous 8-float chunk via two B128 loads.
// ---------------------------------------------------------------------------
__global__ void k_argmax_hist(const float* __restrict__ ref,
                              int* __restrict__ idx, int* __restrict__ cnt) {
  int bq = blockIdx.x;                       // 0 .. B*NQ-1
  const float4* row = (const float4*)(ref + (size_t)bq * HW);
  int l = threadIdx.x;                       // 0..31
  float4 c0 = row[l * 2];
  float4 c1 = row[l * 2 + 1];
  float v[8] = {c0.x, c0.y, c0.z, c0.w, c1.x, c1.y, c1.z, c1.w};
  float bv = -3.402823466e38f;
  int   bi = 0;
  int base = l * 8;                          // contiguous chunk -> first-max OK
  #pragma unroll
  for (int j = 0; j < 8; ++j) {
    if (v[j] > bv) { bv = v[j]; bi = base + j; }
  }
  // wave32 reduce, lower index wins ties
  #pragma unroll
  for (int off = 16; off >= 1; off >>= 1) {
    float ov = __shfl_xor(bv, off, 32);
    int   oi = __shfl_xor(bi, off, 32);
    if (ov > bv || (ov == bv && oi < bi)) { bv = ov; bi = oi; }
  }
  if (l == 0) {
    idx[bq] = bi;
    atomicAdd(&cnt[(bq / NQ) * 256 + bi], 1);
  }
}

// ---------------------------------------------------------------------------
// K2: per-batch mean distance via 256x256 cell-pair counts; block 0 also
//     folds w2 -> v[e]=mean_h w2[e,h] and c=mean(b2).
// ---------------------------------------------------------------------------
__global__ void k_stats(const int* __restrict__ cnt, const float* __restrict__ w2,
                        const float* __restrict__ b2, float* __restrict__ meanout,
                        float* __restrict__ vbar, float* __restrict__ cmean) {
  int bb = blockIdx.x;
  int t  = threadIdx.x;                      // 0..255
  if (bb == 0) {
    float s = 0.f;
    #pragma unroll
    for (int h = 0; h < NH; ++h) s += w2[t * NH + h];
    vbar[t] = s * (1.0f / NH);
    if (t == 0) {
      float sb = 0.f;
      for (int h = 0; h < NH; ++h) sb += b2[h];
      cmean[0] = sb * (1.0f / NH);
    }
  }
  const int* c = cnt + bb * 256;
  int ci = c[t];
  float s = 0.f;
  if (ci) {
    int xi = t & 15, yi = t >> 4;
    for (int j = 0; j < 256; ++j) {
      int cj = c[j];
      if (!cj) continue;
      float dx = (float)(xi - (j & 15));
      float dy = (float)(yi - (j >> 4));
      s += (float)(ci * cj) * sqrtf(dx * dx + dy * dy);
    }
  }
  __shared__ float red[256];
  red[t] = s;
  __syncthreads();
  for (int stride = 128; stride >= 1; stride >>= 1) {
    if (t < stride) red[t] += red[t + stride];
    __syncthreads();
  }
  if (t == 0)
    meanout[bb] = red[0] * (1.0f / 15.0f) / ((float)NQ * (float)NQ);
}

// ---------------------------------------------------------------------------
// K3: LUT[bb][i] = sum_e relu(t_i*w1[e]+b1[e]) * v[e] + c   via WMMA.
//     One full wave32 per (batch, 16-row tile). t_i from bucket i=(dx*16+dy).
//     A: f16 16x32 fragment (relu MLP hidden), B: v replicated over 16 cols,
//     8 unrolled K-steps of v_wmma_f32_16x16x32_f16 contract E=256.
// ---------------------------------------------------------------------------
__global__ void k_lut_wmma(const float* __restrict__ w1, const float* __restrict__ b1,
                           const float* __restrict__ vbar, const float* __restrict__ cmean,
                           const float* __restrict__ mean, float* __restrict__ lut) {
  int tile = blockIdx.x;                     // 0..15
  int bb   = blockIdx.y;                     // batch
  int l    = threadIdx.x;                    // 0..31, EXEC all ones
  int m    = l & 15;                         // A-matrix row for this lane
  int i    = tile * 16 + m;                  // bucket id
  float dxv = (float)(i >> 4), dyv = (float)(i & 15);
  float tval = (sqrtf(dxv * dxv + dyv * dyv) * (1.0f / 15.0f)) / (mean[bb] + 1e-6f);

  int grp8  = (l >> 4) << 3;                 // 0 (lanes 0-15) or 8 (lanes 16-31)
  int bbase = (l >> 4) << 4;                 // B rows: K base 0 or 16

  v8f acc = {};
  #pragma unroll
  for (int k0 = 0; k0 < E; k0 += 32) {
    v16h a, bf;
    #pragma unroll
    for (int h = 0; h < 16; ++h) {
      // 16-bit A 16x32 lane map: h<8 -> K=grp8+h ; h>=8 -> K=16+grp8+(h-8)
      int kA = k0 + ((h < 8) ? (grp8 + h) : (8 + grp8 + h));
      float av = fmaf(tval, w1[kA], b1[kA]);
      a[h]  = (_Float16)fmaxf(av, 0.0f);
      // B 32x16: same K map as A; value independent of column N
      bf[h] = (_Float16)vbar[k0 + ((h < 8) ? (grp8 + h) : (8 + grp8 + h))];
    }
    acc = __builtin_amdgcn_wmma_f32_16x16x32_f16(false, a, false, bf,
                                                 (short)0, acc, false, false);
  }
  float c = cmean[0];
  // D layout: VGPR r -> M=r (lanes 0-15) / M=r+8 (lanes 16-31); N = lane%16
  if (l == 0) {
    #pragma unroll
    for (int r = 0; r < 8; ++r) lut[bb * 256 + tile * 16 + r] = acc[r] + c;
  } else if (l == 16) {
    #pragma unroll
    for (int r = 0; r < 8; ++r) lut[bb * 256 + tile * 16 + 8 + r] = acc[r] + c;
  }
}

// ---------------------------------------------------------------------------
// K4: streaming apply: out[b,q,k] = attn[b,q,k] + lam * LUT[b][bucket(q,k)]
//     One block per (b,q) row. Rows are 900 floats = 3600 B (16B aligned),
//     so attn/out move as B128 float4 and idx as int4. Pure BW-bound:
//     ~13 MB @ 23.3 TB/s ~= 0.56 us.
// ---------------------------------------------------------------------------
__global__ void k_apply(const float* __restrict__ attn, const int* __restrict__ idx,
                        const float* __restrict__ lut, const float* __restrict__ lam_p,
                        float* __restrict__ out) {
  int q  = blockIdx.x;
  int bb = blockIdx.y;
  __shared__ int   sidx[NQ];                 // 900 = 225 int4
  __shared__ float slut[256];
  int t = threadIdx.x;                       // 0..255
  const int4* gidx = (const int4*)(idx + bb * NQ);
  if (t < 225) ((int4*)sidx)[t] = gidx[t];
  slut[t] = lut[bb * 256 + t];
  __syncthreads();
  float lam = lam_p[0];
  int iq = sidx[q];
  int xq = iq & 15, yq = iq >> 4;
  size_t rowoff = ((size_t)bb * NQ + q) * NQ;
  const float4* arow = (const float4*)(attn + rowoff);
  float4*       orow = (float4*)(out + rowoff);
  for (int g = t; g < 225; g += 256) {       // one iter for t < 225
    int4   ik = ((const int4*)sidx)[g];
    float4 a  = arow[g];
    int adx, ady;
    adx = xq - (ik.x & 15); adx = adx < 0 ? -adx : adx;
    ady = yq - (ik.x >> 4); ady = ady < 0 ? -ady : ady;
    a.x += lam * slut[ady * 16 + adx];
    adx = xq - (ik.y & 15); adx = adx < 0 ? -adx : adx;
    ady = yq - (ik.y >> 4); ady = ady < 0 ? -ady : ady;
    a.y += lam * slut[ady * 16 + adx];
    adx = xq - (ik.z & 15); adx = adx < 0 ? -adx : adx;
    ady = yq - (ik.z >> 4); ady = ady < 0 ? -ady : ady;
    a.z += lam * slut[ady * 16 + adx];
    adx = xq - (ik.w & 15); adx = adx < 0 ? -adx : adx;
    ady = yq - (ik.w >> 4); ady = ady < 0 ? -ady : ady;
    a.w += lam * slut[ady * 16 + adx];
    orow[g] = a;
  }
}

extern "C" void kernel_launch(void* const* d_in, const int* in_sizes, int n_in,
                              void* d_out, int out_size, void* d_ws, size_t ws_size,
                              hipStream_t stream) {
  const float* attn = (const float*)d_in[0];
  const float* refp = (const float*)d_in[1];
  const float* lam  = (const float*)d_in[2];
  const float* w1   = (const float*)d_in[3];
  const float* b1   = (const float*)d_in[4];
  const float* w2   = (const float*)d_in[5];
  const float* b2   = (const float*)d_in[6];
  float* out = (float*)d_out;

  char*  ws   = (char*)d_ws;
  int*   idx  = (int*)(ws + OFF_IDX);
  int*   cnt  = (int*)(ws + OFF_CNT);
  float* mean = (float*)(ws + OFF_MEAN);
  float* vbar = (float*)(ws + OFF_VBAR);
  float* cm   = (float*)(ws + OFF_CM);
  float* lut  = (float*)(ws + OFF_LUT);

  hipMemsetAsync(cnt, 0, B * 256 * sizeof(int), stream);
  k_argmax_hist<<<dim3(B * NQ), dim3(32),  0, stream>>>(refp, idx, cnt);
  k_stats      <<<dim3(B),      dim3(256), 0, stream>>>(cnt, w2, b2, mean, vbar, cm);
  k_lut_wmma   <<<dim3(16, B),  dim3(32),  0, stream>>>(w1, b1, vbar, cm, mean, lut);
  k_apply      <<<dim3(NQ, B),  dim3(256), 0, stream>>>(attn, idx, lut, lam, out);
}
